// S2VD_24223615550327
// MI455X (gfx1250) — matrix operand order repaired and checked by reading, not compiled, not run on hardware
//
#include <hip/hip_runtime.h>

// ---------------------------------------------------------------------------
// Types for WMMA bf16 path (gfx1250): v_wmma_f32_16x16x32_bf16
// ---------------------------------------------------------------------------
typedef __bf16 v16bf __attribute__((ext_vector_type(16)));
typedef float  v8f   __attribute__((ext_vector_type(8)));
typedef unsigned short u16x8 __attribute__((ext_vector_type(8)));

union FragU { u16x8 h[2]; v16bf v; };

__device__ __forceinline__ unsigned short f32_to_bf16(float f) {
  unsigned int u = __float_as_uint(f);
  u += 0x7FFFu + ((u >> 16) & 1u);           // round-to-nearest-even
  return (unsigned short)(u >> 16);
}

// A fragment: 16x32 bf16 (MxK), row-major source, per ISA 16-bit A layout.
__device__ __forceinline__ v16bf load_a_frag(const unsigned short* __restrict__ A,
                                             int lda, int k0) {
  int lane = threadIdx.x & 31;
  int r = lane & 15, g = lane >> 4;
  const unsigned short* p = A + (size_t)r * lda + k0 + g * 8;
  FragU u;
  u.h[0] = *(const u16x8*)(p);
  u.h[1] = *(const u16x8*)(p + 16);
  return u.v;
}

// B fragment: 32x16 bf16 (KxN) with B[k,n] = W[n,k] (W row-major, ld=K).
__device__ __forceinline__ v16bf load_b_frag(const unsigned short* __restrict__ W,
                                             int ldb, int k0) {
  int lane = threadIdx.x & 31;
  int n = lane & 15, g = lane >> 4;
  const unsigned short* p = W + (size_t)n * ldb + k0 + g * 8;
  FragU u;
  u.h[0] = *(const u16x8*)(p);
  u.h[1] = *(const u16x8*)(p + 16);
  return u.v;
}

#define WMMA_BF16(a, b, c) \
  __builtin_amdgcn_wmma_f32_16x16x32_bf16(false, (a), false, (b), (short)0, (c), false, false)

// C-tile store with fused bias and optional embedding-gather epilogue.
__device__ __forceinline__ void store_tile(float* __restrict__ C, size_t ldc,
                                           int m0, int n0, v8f acc,
                                           const float* __restrict__ bias,
                                           const float* __restrict__ gW, int gLd,
                                           int gCol0, const int* __restrict__ gIdx) {
  int lane = threadIdx.x & 31;
  int nl = lane & 15, g = lane >> 4;
  int n = n0 + nl;
  float badd = bias ? bias[n] : 0.0f;
#pragma unroll
  for (int r = 0; r < 8; ++r) {
    int m = m0 + r + 8 * g;
    float v = acc[r] + badd;
    if (gW) v += gW[(size_t)n * gLd + gCol0 + gIdx[m]];
    C[(size_t)m * ldc + n] = v;
  }
}

// ---------------------------------------------------------------------------
// Register-blocked WMMA GEMM: C[M,N] = A[M,K] * W[N,K]^T + bias (+ gather).
// One wave -> 32x32 macro-tile (2x2 WMMA tiles, 4 accumulators): 16 FLOP/B of
// operand traffic vs 8 for the 1-tile version. M must be a multiple of 32
// (5120 / 64 here); ragged N (6000) handled by a wave-uniform guard.
// ---------------------------------------------------------------------------
__global__ void k_gemm_bf16(const unsigned short* __restrict__ A, int lda,
                            const unsigned short* __restrict__ W, int ldb,
                            float* __restrict__ C, size_t ldc,
                            int M, int N, int K,
                            const float* __restrict__ bias,
                            const float* __restrict__ gW, int gLd, int gCol0,
                            const int* __restrict__ gIdx) {
  int wave = blockIdx.x * 8 + (threadIdx.x >> 5);
  int tilesN2 = (N + 31) >> 5;
  int tilesM2 = M >> 5;
  int total = tilesM2 * tilesN2;
  if (wave >= total) return;                    // wave-uniform: EXEC stays all-1s
  int tm2 = wave / tilesN2, tn2 = wave % tilesN2;
  int m0 = tm2 * 32, n0 = tn2 * 32;
  bool n1ok = (n0 + 16) < N;                    // second N tile exists (uniform)

  const unsigned short* Ap0 = A + (size_t)m0 * lda;
  const unsigned short* Ap1 = Ap0 + (size_t)16 * lda;
  const unsigned short* Wp0 = W + (size_t)n0 * ldb;
  const unsigned short* Wp1 = Wp0 + (size_t)16 * ldb;

  v8f acc00 = {}, acc01 = {}, acc10 = {}, acc11 = {};
  for (int k0 = 0; k0 < K; k0 += 32) {
    v16bf a0 = load_a_frag(Ap0, lda, k0);
    v16bf a1 = load_a_frag(Ap1, lda, k0);
    v16bf b0 = load_b_frag(Wp0, ldb, k0);
    if (k0 + 32 < K)  // speculative L2 prefetch of next B slab
      __builtin_prefetch(Wp0 + (size_t)(threadIdx.x & 15) * ldb + k0 + 32, 0, 1);
    acc00 = WMMA_BF16(a0, b0, acc00);
    acc10 = WMMA_BF16(a1, b0, acc10);
    if (n1ok) {
      v16bf b1 = load_b_frag(Wp1, ldb, k0);
      acc01 = WMMA_BF16(a0, b1, acc01);
      acc11 = WMMA_BF16(a1, b1, acc11);
    }
  }
  store_tile(C, ldc, m0,      n0,      acc00, bias, gW, gLd, gCol0, gIdx);
  store_tile(C, ldc, m0 + 16, n0,      acc10, bias, gW, gLd, gCol0, gIdx);
  if (n1ok) {
    store_tile(C, ldc, m0,      n0 + 16, acc01, bias, gW, gLd, gCol0, gIdx);
    store_tile(C, ldc, m0 + 16, n0 + 16, acc11, bias, gW, gLd, gCol0, gIdx);
  }
}

// ---------------------------------------------------------------------------
// Fused GRU step. The 16x512 bf16 h slab (16 KB, re-read ~768x/block) is
// staged into LDS with global_load_async_to_lds_b128 (ASYNCcnt path, no VGPR
// round-trip), then A-fragments come from LDS (ds_load_b128). gh (16x768 f32)
// also lives in LDS. Total 64 KB static LDS (<< 320 KB/WGP).
// Grid: 8 blocks = 4 row-blocks x 2 column-halves. gi already contains bih.
// ---------------------------------------------------------------------------
__global__ void k_gru_step(const float* __restrict__ gi,             // [64 x 1536]
                           const float* __restrict__ h_in_f,         // [64 x 512]
                           const unsigned short* __restrict__ h_in_b,// [64 x 512] bf16
                           const unsigned short* __restrict__ Whh_b, // [1536 x 512] bf16
                           const float* __restrict__ bhh,            // [1536]
                           float* __restrict__ h_out_f,
                           unsigned short* __restrict__ h_out_b,
                           unsigned short* __restrict__ enc_out) {   // nullable
  __shared__ float gh[16 * 768];                 // 48 KB
  __shared__ unsigned short sh_h[16 * 512];      // 16 KB h stage
  int rowBlk = blockIdx.x >> 1;
  int half   = blockIdx.x & 1;
  int row0 = rowBlk * 16;
  int h0   = half * 256;
  int wave = threadIdx.x >> 5;

  // --- async copy of the 16 KB contiguous h slab into LDS ---
  {
    const unsigned short* src = h_in_b + (size_t)row0 * 512;
    // generic pointer low 32 bits == wave-relative LDS byte offset (ISA 10.2)
    unsigned ldsbase = (unsigned)(size_t)(void*)&sh_h[0];
    for (int i = threadIdx.x; i < 1024; i += 256) {        // 1024 x 16 B
      unsigned dst = ldsbase + i * 16;
      const void* gp = (const void*)(src + i * 8);
      asm volatile("global_load_async_to_lds_b128 %0, %1, off"
                   :: "v"(dst), "v"(gp) : "memory");
    }
    asm volatile("s_wait_asynccnt 0x0" ::: "memory");
  }
  __syncthreads();

  for (int ti = wave; ti < 48; ti += 8) {        // 48 tiles of 16x16 gh columns
    int j   = ti >> 4;                           // gate 0..2
    int cc0 = (ti & 15) << 4;                    // 0..240
    int n0  = j * 512 + h0 + cc0;
    const unsigned short* Wp = Whh_b + (size_t)n0 * 512;
    v8f acc = {};
    for (int k0 = 0; k0 < 512; k0 += 32) {
      v16bf a = load_a_frag(&sh_h[0], 512, k0);  // ds_load_b128 from LDS
      v16bf b = load_b_frag(Wp, 512, k0);
      acc = WMMA_BF16(a, b, acc);
    }
    int lane = threadIdx.x & 31;
    int nl = lane & 15, g = lane >> 4;
#pragma unroll
    for (int r = 0; r < 8; ++r) {
      int m = r + 8 * g;
      gh[m * 768 + j * 256 + cc0 + nl] = acc[r];
    }
  }
  __syncthreads();

  for (int i = threadIdx.x; i < 16 * 256; i += 256) {
    int mr = i >> 8;
    int cc = i & 255;
    int b = row0 + mr;
    int c = h0 + cc;
    float ir  = gi[(size_t)b * 1536 + c];
    float iz  = gi[(size_t)b * 1536 + 512 + c];
    float in_ = gi[(size_t)b * 1536 + 1024 + c];
    float hr = gh[mr * 768 + cc]       + bhh[c];
    float hz = gh[mr * 768 + 256 + cc] + bhh[512 + c];
    float hn = gh[mr * 768 + 512 + cc] + bhh[1024 + c];
    float rg = 1.0f / (1.0f + __expf(-(ir + hr)));
    float zg = 1.0f / (1.0f + __expf(-(iz + hz)));
    float ng = tanhf(in_ + rg * hn);
    float hp = h_in_f[(size_t)b * 512 + c];
    float hv = (1.0f - zg) * ng + zg * hp;
    h_out_f[(size_t)b * 512 + c] = hv;
    unsigned short hb = f32_to_bf16(hv);
    h_out_b[(size_t)b * 512 + c] = hb;
    if (enc_out) enc_out[(size_t)b * 512 + c] = hb;
  }
}

// ---------------------------------------------------------------------------
// Utility kernels
// ---------------------------------------------------------------------------
__global__ void k_convert_bf16(const float* __restrict__ src,
                               unsigned short* __restrict__ dst,
                               int rows, int cols, int src_ld, int src_col0) {
  long long i = (long long)blockIdx.x * blockDim.x + threadIdx.x;
  long long total = (long long)rows * cols;
  if (i >= total) return;
  int r = (int)(i / cols), c = (int)(i % cols);
  dst[i] = f32_to_bf16(src[(size_t)r * src_ld + src_col0 + c]);
}

// x[b,t,f] (B,T,F) -> Xb[(t*B+b)*F + f] bf16
__global__ void k_pack_x(const float* __restrict__ x,
                         unsigned short* __restrict__ xb, int B, int T, int F) {
  long long i = (long long)blockIdx.x * blockDim.x + threadIdx.x;
  if (i >= (long long)B * T * F) return;
  int f = (int)(i % F);
  long long m = i / F;
  int t = (int)(m / B), b = (int)(m % B);
  xb[i] = f32_to_bf16(x[((size_t)b * T + t) * F + f]);
}

__global__ void k_bcast_rows(const float* __restrict__ v, float* __restrict__ out,
                             int rows, int cols) {
  long long i = (long long)blockIdx.x * blockDim.x + threadIdx.x;
  if (i < (long long)rows * cols) out[i] = v[i % cols];
}

__global__ void k_zero32(unsigned int* p, long long n) {
  long long i = (long long)blockIdx.x * blockDim.x + threadIdx.x;
  if (i < n) p[i] = 0u;
}

// probs[:,0,:] = 0 except probs[0,0,1]=1 ; loss slot = 0
__global__ void k_init_out(float* out, int B, int L, int Wd) {
  long long i = (long long)blockIdx.x * blockDim.x + threadIdx.x;
  long long tot = (long long)B * Wd;
  if (i < tot) {
    int b = (int)(i / Wd), w = (int)(i % Wd);
    out[(size_t)b * L * Wd + w] = (b == 0 && w == 1) ? 1.0f : 0.0f;
  }
  if (i == 0) out[(size_t)B * L * Wd] = 0.0f;
}

// loss = -mean_b log_softmax(probs[b, L-1, :])[y[L-1, b]]
__global__ void k_loss(const float* __restrict__ probs, const int* __restrict__ y,
                       float* loss, int B, int L, int Wd) {
  __shared__ float red[256];
  int b = blockIdx.x;
  const float* row = probs + ((size_t)b * L + (L - 1)) * Wd;
  float mx = -3.4e38f;
  for (int w2 = threadIdx.x; w2 < Wd; w2 += 256) mx = fmaxf(mx, row[w2]);
  red[threadIdx.x] = mx;
  __syncthreads();
  for (int s = 128; s > 0; s >>= 1) {
    if ((int)threadIdx.x < s) red[threadIdx.x] = fmaxf(red[threadIdx.x], red[threadIdx.x + s]);
    __syncthreads();
  }
  mx = red[0];
  __syncthreads();
  float sum = 0.0f;
  for (int w2 = threadIdx.x; w2 < Wd; w2 += 256) sum += __expf(row[w2] - mx);
  red[threadIdx.x] = sum;
  __syncthreads();
  for (int s = 128; s > 0; s >>= 1) {
    if ((int)threadIdx.x < s) red[threadIdx.x] += red[threadIdx.x + s];
    __syncthreads();
  }
  if (threadIdx.x == 0) {
    int tgt = y[(size_t)(L - 1) * B + b];
    float lp = (row[tgt] - mx) - __logf(red[0]);
    atomicAdd(loss, -lp / (float)B);
  }
}

// ---------------------------------------------------------------------------
// Host orchestration
// ---------------------------------------------------------------------------
static inline dim3 grid1(long long n) { return dim3((unsigned)((n + 255) / 256)); }

static void launch_gemm(hipStream_t s, const unsigned short* A, int lda,
                        const unsigned short* W, int ldb, float* C, size_t ldc,
                        int M, int N, int K, const float* bias,
                        const float* gW, int gLd, int gCol0, const int* gIdx) {
  int tiles = (M / 32) * ((N + 31) / 32);
  int blocks = (tiles + 7) / 8;
  k_gemm_bf16<<<blocks, 256, 0, s>>>(A, lda, W, ldb, C, ldc, M, N, K,
                                     bias, gW, gLd, gCol0, gIdx);
}

extern "C" void kernel_launch(void* const* d_in, const int* in_sizes, int n_in,
                              void* d_out, int out_size, void* d_ws, size_t ws_size,
                              hipStream_t stream) {
  (void)in_sizes; (void)n_in; (void)out_size; (void)ws_size;
  const float* x     = (const float*)d_in[0];
  const int*   y     = (const int*)  d_in[1];
  const float* Wih_v = (const float*)d_in[2];
  const float* Whh_v = (const float*)d_in[3];
  const float* bih_v = (const float*)d_in[4];
  const float* bhh_v = (const float*)d_in[5];
  const float* Wih_c = (const float*)d_in[6];
  const float* Whh_c = (const float*)d_in[7];
  const float* bih_c = (const float*)d_in[8];
  const float* bhh_c = (const float*)d_in[9];
  const float* Wout  = (const float*)d_in[10];
  const float* bout  = (const float*)d_in[11];
  float* out = (float*)d_out;

  const int B = 64, T = 80, F = 4096, Wd = 6000, H = 512, G = 1536, L = 32;
  const int Wc = H + Wd;          // 6512
  const int M1 = T * B;           // 5120

  // workspace layout (~104 MB)
  char* w = (char*)d_ws;
  size_t off = 0;
  auto alloc = [&](size_t bytes) -> void* {
    void* p = w + off;
    off = (off + bytes + 255) & ~(size_t)255;
    return p;
  };
  unsigned short* Xb    = (unsigned short*)alloc((size_t)M1 * F * 2);
  unsigned short* Wihvb = (unsigned short*)alloc((size_t)G * F * 2);
  unsigned short* Whhvb = (unsigned short*)alloc((size_t)G * H * 2);
  unsigned short* Wichb = (unsigned short*)alloc((size_t)G * H * 2);
  unsigned short* Whhcb = (unsigned short*)alloc((size_t)G * H * 2);
  unsigned short* Woutb = (unsigned short*)alloc((size_t)Wd * H * 2);
  float*          giBuf = (float*)alloc((size_t)M1 * G * 4);      // gi_v then gi_c
  unsigned short* encb  = (unsigned short*)alloc((size_t)M1 * H * 2);
  float*          giZ   = (float*)alloc((size_t)B * G * 4);
  float*          giD   = (float*)alloc((size_t)B * G * 4);
  float*          hAf   = (float*)alloc((size_t)B * H * 4);
  float*          hBf   = (float*)alloc((size_t)B * H * 4);
  unsigned short* hAb   = (unsigned short*)alloc((size_t)B * H * 2);
  unsigned short* hBb   = (unsigned short*)alloc((size_t)B * H * 2);
  float*          hcf   = (float*)alloc((size_t)B * H * 4);
  unsigned short* hcb   = (unsigned short*)alloc((size_t)B * H * 2);

  // --- one-time conversions / init ---
  k_convert_bf16<<<grid1((long long)G * F), 256, 0, stream>>>(Wih_v, Wihvb, G, F, F, 0);
  k_convert_bf16<<<grid1((long long)G * H), 256, 0, stream>>>(Whh_v, Whhvb, G, H, H, 0);
  k_convert_bf16<<<grid1((long long)G * H), 256, 0, stream>>>(Wih_c, Wichb, G, H, Wc, 0);
  k_convert_bf16<<<grid1((long long)G * H), 256, 0, stream>>>(Whh_c, Whhcb, G, H, H, 0);
  k_convert_bf16<<<grid1((long long)Wd * H), 256, 0, stream>>>(Wout, Woutb, Wd, H, H, 0);
  k_pack_x<<<grid1((long long)M1 * F), 256, 0, stream>>>(x, Xb, B, T, F);
  k_bcast_rows<<<grid1((long long)B * G), 256, 0, stream>>>(bih_v, giZ, B, G);
  k_zero32<<<grid1(B * H), 256, 0, stream>>>((unsigned int*)hAf, B * H);
  k_zero32<<<grid1(B * H / 2), 256, 0, stream>>>((unsigned int*)hAb, B * H / 2);
  k_init_out<<<grid1((long long)B * Wd), 256, 0, stream>>>(out, B, L, Wd);

  // --- gi_v = Xb @ Wih_v^T + bih_v : 5120 x 4096 x 1536 (64 GFLOP, WMMA) ---
  launch_gemm(stream, Xb, F, Wihvb, F, giBuf, (size_t)G, M1, G, F, bih_v,
              nullptr, 0, 0, nullptr);

  // --- encoder-v scan (80 fused GRU steps) ---
  float* curf = hAf; unsigned short* curb = hAb;
  float* nxtf = hBf; unsigned short* nxtb = hBb;
  for (int t = 0; t < T; ++t) {
    k_gru_step<<<8, 256, 0, stream>>>(giBuf + (size_t)t * B * G, curf, curb,
                                      Whhvb, bhh_v, nxtf, nxtb,
                                      encb + (size_t)t * B * H);
    float* tf = curf; curf = nxtf; nxtf = tf;
    unsigned short* tb = curb; curb = nxtb; nxtb = tb;
  }

  // --- gi_c = enc_out @ Wih_c_h^T + bih_c : 5120 x 512 x 1536 ---
  launch_gemm(stream, encb, H, Wichb, H, giBuf, (size_t)G, M1, G, H, bih_c,
              nullptr, 0, 0, nullptr);

  // --- encoder-c scan ---
  k_zero32<<<grid1(B * H), 256, 0, stream>>>((unsigned int*)curf, B * H);
  k_zero32<<<grid1(B * H / 2), 256, 0, stream>>>((unsigned int*)curb, B * H / 2);
  for (int t = 0; t < T; ++t) {
    k_gru_step<<<8, 256, 0, stream>>>(giBuf + (size_t)t * B * G, curf, curb,
                                      Whhcb, bhh_c, nxtf, nxtb, nullptr);
    float* tf = curf; curf = nxtf; nxtf = tf;
    unsigned short* tb = curb; curb = nxtb; nxtb = tb;
  }
  // hidden_video now in (curf, curb)

  // --- decoder: 31 steps ---
  for (int t = 0; t < L - 1; ++t) {
    // h_v2 = GRU(gi_zero=bih_v, h_v, Whh_v, bhh_v)
    k_gru_step<<<8, 256, 0, stream>>>(giZ, curf, curb, Whhvb, bhh_v,
                                      nxtf, nxtb, nullptr);
    // gi = h_v2 @ Wih_c_h^T + Wih_c_w[:, y[t]]^T + bih_c
    launch_gemm(stream, nxtb, H, Wichb, H, giD, (size_t)G, B, G, H, bih_c,
                Wih_c, Wc, H, y + (size_t)t * B);
    // h_c = GRU(gi, h_v2, Whh_c, bhh_c)
    k_gru_step<<<8, 256, 0, stream>>>(giD, nxtf, nxtb, Whhcb, bhh_c,
                                      hcf, hcb, nullptr);
    // logits -> probs[:, t+1, :]
    launch_gemm(stream, hcb, H, Woutb, H, out + (size_t)(t + 1) * Wd,
                (size_t)L * Wd, B, Wd, H, bout, nullptr, 0, 0, nullptr);
    // h_v <- h_v2
    float* tf = curf; curf = nxtf; nxtf = tf;
    unsigned short* tb = curb; curb = nxtb; nxtb = tb;
  }

  // --- loss from probs[:, L-1, :] ---
  k_loss<<<B, 256, 0, stream>>>(out, y, out + (size_t)B * L * Wd, B, L, Wd);
}
